// RegularNeuralField_17154099380948
// MI455X (gfx1250) — matrix operand order repaired and checked by reading, not compiled, not run on hardware
//
#include <hip/hip_runtime.h>

typedef __attribute__((ext_vector_type(16))) __bf16    v16bf;
typedef __attribute__((ext_vector_type(2)))  __bf16    bf16x2;
typedef __attribute__((ext_vector_type(8)))  float     v8f;
typedef __attribute__((ext_vector_type(4)))  float     f32x4;
typedef __attribute__((ext_vector_type(2)))  float     f32x2;
typedef __attribute__((ext_vector_type(4)))  unsigned  u32x4;
typedef __attribute__((ext_vector_type(2)))  unsigned  u32x2;

#define GH    512
#define GW    512
#define FD    64
#define HIDN  256
#define NPTS  (1u << 20)
#define WAVES 8
#define TPW   8          // wave-tiles per wave per block
#define MPT   32         // points per wave-tile: two 16-point B-tiles
#define FSTR  72         // bf16 per feat row (64 + pad)
#define KSTR  264        // bf16 per act row (256 + pad) -> bank step 4/row
#define WBYTES  174144   // fW1(32768) + fW2(131072) + fW3(8192) + fB(2112)
#define WCHUNKS (WBYTES / 16)                 // 10884 x 16B
#define ACTB    (MPT * KSTR * 2)              // 16896 B per wave
#define SMEMB   (WBYTES + WAVES * ACTB)       // 309312 B <= 320KB WGP LDS

// ---------------- helpers ----------------

__device__ __forceinline__ unsigned short f2bf_rne(float f) {   // prep only
  union { float f; unsigned u; } x; x.f = f;
  unsigned r = x.u + 0x7FFFu + ((x.u >> 16) & 1u);
  return (unsigned short)(r >> 16);
}

// one v_cvt_pk_bf16_f32
__device__ __forceinline__ unsigned cvt_pack2(float a, float b) {
  f32x2 t = {a, b};
  return __builtin_bit_cast(unsigned, __builtin_convertvector(t, bf16x2));
}

// one v_cvt_pk_bf16_f32 + one v_pk_max_num_bf16 (relu in bf16 domain)
__device__ __forceinline__ unsigned relu_pack2(float a, float b) {
  f32x2 t = {a, b};
  bf16x2 p = __builtin_convertvector(t, bf16x2);
  bf16x2 z = {};
  p = __builtin_elementwise_max(p, z);
  return __builtin_bit_cast(unsigned, p);
}

__device__ __forceinline__ u32x4 relu_pack8(const v8f& c) {
  u32x4 s;
  s.x = relu_pack2(c[0], c[1]);
  s.y = relu_pack2(c[2], c[3]);
  s.z = relu_pack2(c[4], c[5]);
  s.w = relu_pack2(c[6], c[7]);
  return s;
}

// A-fragment (16-bit MxK 16x32) swizzled index for weights stored as W^T tiles:
// element (k, n): lane = ((k>>4)&1)*16 + (n&15), pos = (k&15)
__device__ __forceinline__ int afragIdx(int k, int n, int numKc) {
  int nt = n >> 4, kc = k >> 5;
  int kp = k & 31;
  int h = kp >> 4;
  int lane = h * 16 + (n & 15);
  int pos = kp & 15;
  return ((nt * numKc + kc) * 32 + lane) * 16 + pos;
}

// ---------------- weight prep: f32 -> bf16, swizzle to A-fragment order ----------------

__global__ void prep_weights(const float* __restrict__ W1, const float* __restrict__ b1,
                             const float* __restrict__ W2, const float* __restrict__ b2,
                             const float* __restrict__ W3, const float* __restrict__ b3,
                             unsigned short* __restrict__ fW1,
                             unsigned short* __restrict__ fW2,
                             unsigned short* __restrict__ fW3,
                             float* __restrict__ fB) {
  int tid = blockIdx.x * blockDim.x + threadIdx.x;
  int stride = gridDim.x * blockDim.x;
  for (int i = tid; i < FD * HIDN; i += stride) {            // W1 [k=64][n=256]
    int k = i >> 8, n = i & 255;
    fW1[afragIdx(k, n, 2)] = f2bf_rne(W1[i]);
  }
  for (int i = tid; i < HIDN * HIDN; i += stride) {          // W2 [k=256][n=256]
    int k = i >> 8, n = i & 255;
    fW2[afragIdx(k, n, 8)] = f2bf_rne(W2[i]);
  }
  for (int i = tid; i < HIDN * 16; i += stride) {            // W3 [k=256][n=3->16]
    int k = i >> 4, n = i & 15;
    float v = (n < 3) ? W3[k * 3 + n] : 0.0f;
    fW3[afragIdx(k, n, 8)] = f2bf_rne(v);
  }
  for (int i = tid; i < 528; i += stride) {                  // b1[256] b2[256] b3 pad16
    float v;
    if (i < 256)      v = b1[i];
    else if (i < 512) v = b2[i - 256];
    else              v = (i - 512 < 3) ? b3[i - 512] : 0.0f;
    fB[i] = v;
  }
}

// ---------------- main fused kernel ----------------

union Frag { v16bf v; u32x4 q[2]; };

#define WMMA_BF16(A, B, C) \
  __builtin_amdgcn_wmma_f32_16x16x32_bf16(false, (A), false, (B), (short)0, (C), false, false)

__device__ __forceinline__ v8f loadBias8(const float* p) {    // 8 consecutive f32
  f32x4 lo = *(const f32x4*)p;
  f32x4 hi = *(const f32x4*)(p + 4);
  return __builtin_shufflevector(lo, hi, 0, 1, 2, 3, 4, 5, 6, 7);
}

__global__ __launch_bounds__(256) void nf_kernel(const float* __restrict__ coords,
                                                 const float* __restrict__ grid,
                                                 const unsigned char* __restrict__ wsbase,
                                                 float* __restrict__ out) {
  extern __shared__ unsigned char smem[];
  unsigned short* sW1 = (unsigned short*)smem;               // 16384 bf16
  unsigned short* sW2 = sW1 + 16384;                         // 65536 bf16
  unsigned short* sW3 = sW2 + 65536;                         // 4096 bf16
  float*          sB  = (float*)(sW3 + 4096);                // 528 f32
  unsigned char*  sActBase = smem + WBYTES;

  const int lane = threadIdx.x & 31;
  const int wave = threadIdx.x >> 5;
  const int pt   = lane & 15;       // point within 16-pt B-tile == D column (N)
  const int hh   = lane >> 4;       // lane half: K-half for B, M-half for D

  unsigned short* myAct = (unsigned short*)(sActBase + wave * ACTB); // [pt][k] rows
  const u32x4* w1q = (const u32x4*)sW1;
  const u32x4* w2q = (const u32x4*)sW2;
  const u32x4* w3q = (const u32x4*)sW3;

  // ---- async stage of weights+biases: global -> LDS (ASYNCcnt path) ----
  {
    unsigned ldsBase = (unsigned)(size_t)(void*)smem;
    unsigned long long gbase = (unsigned long long)wsbase;
    for (int i = threadIdx.x; i < WCHUNKS; i += 256) {
      unsigned off = (unsigned)i * 16u;
      unsigned lds = ldsBase + off;
      asm volatile("global_load_async_to_lds_b128 %0, %1, %2 offset:0"
                   :: "v"(lds), "v"(off), "s"(gbase) : "memory");
    }
  }

  // ---- bilinear gather: 1 lane = 1 point, row-major [pt][64] (natural B layout) ----
  auto gather = [&](int ptBase) {
    const int p = ptBase + lane;
    const float cx = coords[p * 2 + 0];
    const float cy = coords[p * 2 + 1];
    const float x = (cx * 0.5f + 0.5f) * (float)(GW - 1);
    const float y = (cy * 0.5f + 0.5f) * (float)(GH - 1);
    int x0 = (int)floorf(x); x0 = x0 < 0 ? 0 : (x0 > GW - 2 ? GW - 2 : x0);
    int y0 = (int)floorf(y); y0 = y0 < 0 ? 0 : (y0 > GH - 2 ? GH - 2 : y0);
    const float tx = x - (float)x0, ty = y - (float)y0;
    const float w00 = (1.f - tx) * (1.f - ty), w01 = tx * (1.f - ty);
    const float w10 = (1.f - tx) * ty,         w11 = tx * ty;
    const float* g00 = grid + ((size_t)y0 * GW + (size_t)x0) * FD;
    const f32x4* p00 = (const f32x4*)g00;
    const f32x4* p01 = (const f32x4*)(g00 + FD);
    const f32x4* p10 = (const f32x4*)(g00 + (size_t)GW * FD);
    const f32x4* p11 = (const f32x4*)(g00 + (size_t)GW * FD + FD);
    unsigned* frow = (unsigned*)(myAct + lane * FSTR);
#pragma unroll
    for (int j = 0; j < 16; ++j) {
      f32x4 a = p00[j], b = p01[j], c = p10[j], d = p11[j];
      f32x4 r = a * w00 + b * w01 + c * w10 + d * w11;
      u32x2 pk;
      pk.x = cvt_pack2(r.x, r.y);
      pk.y = cvt_pack2(r.z, r.w);
      *(u32x2*)(frow + j * 2) = pk;
    }
  };

  const int tile0 = (blockIdx.x * WAVES + wave) * TPW;

  gather(tile0 * MPT);                              // overlap with async weight DMA
  asm volatile("s_wait_asynccnt 0x0" ::: "memory");
  __syncthreads();

  for (int t = 0; t < TPW; ++t) {
    const int ptBase = (tile0 + t) * MPT;
    if (t > 0) gather(ptBase);

    // ---- layer 1: A=W1^T tiles, B=feats [64k x 16pt] x2 tiles ----
    Frag f1[2][2];                                  // preload B (feats) -> regs
#pragma unroll
    for (int tt = 0; tt < 2; ++tt)
#pragma unroll
      for (int kc = 0; kc < 2; ++kc) {
        const u32x4* p = (const u32x4*)(myAct + (tt * 16 + pt) * FSTR + kc * 32 + hh * 16);
        f1[tt][kc].q[0] = p[0];
        f1[tt][kc].q[1] = p[1];
      }
#pragma unroll 2
    for (int nt = 0; nt < 16; ++nt) {
      v8f cLo = loadBias8(sB + nt * 16 + hh * 8);   // bias along M (channels)
      v8f cHi = cLo;
      Frag w0, w1;
      w0.q[0] = w1q[((nt * 2 + 0) * 32 + lane) * 2 + 0];
      w0.q[1] = w1q[((nt * 2 + 0) * 32 + lane) * 2 + 1];
      w1.q[0] = w1q[((nt * 2 + 1) * 32 + lane) * 2 + 0];
      w1.q[1] = w1q[((nt * 2 + 1) * 32 + lane) * 2 + 1];
      cLo = WMMA_BF16(w0.v, f1[0][0].v, cLo);
      cHi = WMMA_BF16(w0.v, f1[1][0].v, cHi);
      cLo = WMMA_BF16(w1.v, f1[0][1].v, cLo);
      cHi = WMMA_BF16(w1.v, f1[1][1].v, cHi);
      // D lane: point=pt, channels nt*16+hh*8+(0..7) -> 16B contiguous store
      *(u32x4*)(myAct + (pt)      * KSTR + nt * 16 + hh * 8) = relu_pack8(cLo);
      *(u32x4*)(myAct + (16 + pt) * KSTR + nt * 16 + hh * 8) = relu_pack8(cHi);
    }

    // ---- layer 2: A=W2^T, B=h1 [256k x 16pt] x2; preload B so act can be reused ----
    Frag h1f[2][8];
#pragma unroll
    for (int tt = 0; tt < 2; ++tt)
#pragma unroll
      for (int kc = 0; kc < 8; ++kc) {
        const u32x4* p = (const u32x4*)(myAct + (tt * 16 + pt) * KSTR + kc * 32 + hh * 16);
        h1f[tt][kc].q[0] = p[0];
        h1f[tt][kc].q[1] = p[1];
      }
#pragma unroll 2
    for (int nt = 0; nt < 16; ++nt) {
      v8f cLo = loadBias8(sB + 256 + nt * 16 + hh * 8);
      v8f cHi = cLo;
#pragma unroll
      for (int kc = 0; kc < 8; ++kc) {
        Frag w;
        w.q[0] = w2q[((nt * 8 + kc) * 32 + lane) * 2 + 0];
        w.q[1] = w2q[((nt * 8 + kc) * 32 + lane) * 2 + 1];
        cLo = WMMA_BF16(w.v, h1f[0][kc].v, cLo);
        cHi = WMMA_BF16(w.v, h1f[1][kc].v, cHi);
      }
      *(u32x4*)(myAct + (pt)      * KSTR + nt * 16 + hh * 8) = relu_pack8(cLo);
      *(u32x4*)(myAct + (16 + pt) * KSTR + nt * 16 + hh * 8) = relu_pack8(cHi);
    }

    // ---- layer 3: A=W3^T [16(3) x 256k] hoisted, B=h2 per point-tile ----
    {
      Frag w3f[8];
#pragma unroll
      for (int kc = 0; kc < 8; ++kc) {
        w3f[kc].q[0] = w3q[(kc * 32 + lane) * 2 + 0];
        w3f[kc].q[1] = w3q[(kc * 32 + lane) * 2 + 1];
      }
      v8f c0 = loadBias8(sB + 512 + hh * 8);
      v8f c1 = c0;
#pragma unroll
      for (int kc = 0; kc < 8; ++kc) {
        Frag b0, b1;
        const u32x4* p0 = (const u32x4*)(myAct + (pt)      * KSTR + kc * 32 + hh * 16);
        const u32x4* p1 = (const u32x4*)(myAct + (16 + pt) * KSTR + kc * 32 + hh * 16);
        b0.q[0] = p0[0]; b0.q[1] = p0[1];
        b1.q[0] = p1[0]; b1.q[1] = p1[1];
        c0 = WMMA_BF16(w3f[kc].v, b0.v, c0);
        c1 = WMMA_BF16(w3f[kc].v, b1.v, c1);
      }
      if (hh == 0) {                                 // channels 0..2 live in half 0
        float* o0 = out + (size_t)(ptBase + pt) * 3;
        float* o1 = out + (size_t)(ptBase + 16 + pt) * 3;
        o0[0] = c0[0]; o0[1] = c0[1]; o0[2] = c0[2];
        o1[0] = c1[0]; o1[1] = c1[1]; o1[2] = c1[2];
      }
    }
  }
}

// ---------------- launch ----------------

extern "C" void kernel_launch(void* const* d_in, const int* in_sizes, int n_in,
                              void* d_out, int out_size, void* d_ws, size_t ws_size,
                              hipStream_t stream) {
  (void)in_sizes; (void)n_in; (void)out_size; (void)ws_size;
  const float* coords = (const float*)d_in[0];
  const float* grid   = (const float*)d_in[1];
  const float* W1 = (const float*)d_in[2];
  const float* b1 = (const float*)d_in[3];
  const float* W2 = (const float*)d_in[4];
  const float* b2 = (const float*)d_in[5];
  const float* W3 = (const float*)d_in[6];
  const float* b3 = (const float*)d_in[7];
  float* out = (float*)d_out;

  unsigned short* fW1 = (unsigned short*)d_ws;      // workspace layout == LDS layout
  unsigned short* fW2 = fW1 + 16384;
  unsigned short* fW3 = fW2 + 65536;
  float*          fB  = (float*)(fW3 + 4096);

  prep_weights<<<128, 256, 0, stream>>>(W1, b1, W2, b2, W3, b3, fW1, fW2, fW3, fB);

  hipFuncSetAttribute((const void*)nf_kernel,
                      hipFuncAttributeMaxDynamicSharedMemorySize, (int)SMEMB);

  const unsigned blocks = NPTS / (WAVES * TPW * MPT);   // 512
  nf_kernel<<<blocks, 256, SMEMB, stream>>>(coords, grid,
                                            (const unsigned char*)d_ws, out);
}
